// OutputModule_87196426044007
// MI455X (gfx1250) — matrix operand order repaired and checked by reading, not compiled, not run on hardware
//
#include <hip/hip_runtime.h>
#include <math.h>

typedef __attribute__((ext_vector_type(16))) _Float16 v16h;
typedef __attribute__((ext_vector_type(8)))  float    v8f;

#define D_   512
#define FF_  1024
#define K3D  1536          // 3*D
#define B_   16
#define NPG_ 512
#define EPG_ 1536
#define S_   2048
#define NN_  8192
#define NE_  24576
#define NT_  64            // FF/16 n-tiles
#define KT_  48            // 3D/32 k-tiles

// ---------------------------------------------------------------- zero out
__global__ void zero_out_kernel(float* __restrict__ out, int n) {
    int i = blockIdx.x * blockDim.x + threadIdx.x;
    if (i < n) out[i] = 0.0f;
}

// ---------------------------------------------------------------- pack W1 -> f16 fragments
// Fragment layout (per 16x32 B-tile): 32 lanes x 16 halves contiguous.
// B 32x16, this file's contiguous k order (same as A): lane L -> n = L&15 ; k = kt*32 + ((L>=16)?16:0) + h  (h = 0..15)
__global__ __launch_bounds__(256)
void pack_w1_kernel(const float* __restrict__ W1e,
                    const float* __restrict__ W1f,
                    _Float16* __restrict__ Wpk) {
    int tid  = blockIdx.x * 256 + threadIdx.x;   // 2*64*48*32 = 196608 threads
    int lane = tid & 31;
    int frag = tid >> 5;                          // 0..6143
    int mlp  = frag / (NT_ * KT_);
    int rem  = frag - mlp * (NT_ * KT_);
    int nt   = rem / KT_;
    int kt   = rem - nt * KT_;
    const float* W1 = mlp ? W1f : W1e;
    int n     = nt * 16 + (lane & 15);
    int kbase = kt * 32 + ((lane >= 16) ? 16 : 0);
    _Float16* out = Wpk + (size_t)frag * 512 + lane * 16;
    #pragma unroll
    for (int h = 0; h < 16; ++h) {
        out[h] = (_Float16)W1[(kbase + h) * FF_ + n];
    }
}

// ---------------------------------------------------------------- fused edge MLP
__global__ __launch_bounds__(256)
void edge_mlp_kernel(const float* __restrict__ x,
                     const float* __restrict__ pos,
                     const int*   __restrict__ ei,   // [2, NE]
                     const _Float16* __restrict__ Wpk,
                     const float* __restrict__ b1e,
                     const float* __restrict__ W2e,
                     const float* __restrict__ b2e,
                     const float* __restrict__ b1f,
                     const float* __restrict__ W2f,
                     const float* __restrict__ b2f,
                     float* __restrict__ out) {      // [16 energy][NN*3 forces]
    __shared__ _Float16 sA[16 * K3D];                // 48KB, fragment-major A tiles
    __shared__ float    sRed[8][16][2];
    __shared__ int      sSrc[16], sDst[16];
    __shared__ float    sE2[16];

    const int blk  = blockIdx.x;                     // 1536 blocks, 16 edges each
    const int tid  = threadIdx.x;
    const int wave = tid >> 5;
    const int lane = tid & 31;

    if (tid < 16) {
        int e = blk * 16 + tid;
        sSrc[tid] = ei[e];
        sDst[tid] = ei[NE_ + e];
    }
    __syncthreads();

    // ---- gather inputs [16, 1536] from x into LDS, pre-swizzled to WMMA A layout.
    // A 16x32, this file's contiguous k order (same as B pack): lane = m + 16*(kin>>4) ; half h = kin&15
    for (int i = tid; i < 16 * K3D; i += 256) {
        int m = i / K3D;
        int c = i - m * K3D;
        int e = blk * 16 + m;
        float v;
        if (c < 1024) {
            int node = (c < D_) ? sSrc[m] : sDst[m];
            int d    = c & (D_ - 1);
            int s    = node & (NPG_ - 1);            // local node -> row s
            int b    = node >> 9;                    // graph
            v = x[((s * B_) + b) * D_ + d];
        } else {
            int d = c - 1024;
            int g = e / EPG_;
            int t = e - g * EPG_;
            v = x[(((NPG_ + t) * B_) + g) * D_ + d];
        }
        int kt     = c >> 5;
        int kin    = c & 31;
        int laneHi = (kin >> 4) & 1;
        int h      = kin & 15;
        sA[kt * 512 + (m + (laneHi << 4)) * 16 + h] = (_Float16)v;
    }
    __syncthreads();

    // ---- two fused GEMMs: acc over K=1536, epilogue GELU * W2 -> per-edge scalar
    float scal[2][8] = {};
    #pragma unroll
    for (int mlp = 0; mlp < 2; ++mlp) {
        const _Float16* Bbase = Wpk + (size_t)mlp * NT_ * KT_ * 512;
        const float*    b1    = mlp ? b1f : b1e;
        const float*    W2    = mlp ? W2f : W2e;
        for (int nt = wave; nt < NT_; nt += 8) {     // 8 n-tiles per wave, uniform
            const _Float16* bp = Bbase + (size_t)nt * KT_ * 512 + lane * 16;
            v8f acc = {};
            v16h aCur = *(const v16h*)(&sA[lane * 16]);
            v16h bCur = *(const v16h*)(bp);
            for (int kt = 0; kt < KT_ - 1; ++kt) {
                v16h aNxt = *(const v16h*)(&sA[(kt + 1) * 512 + lane * 16]);
                v16h bNxt = *(const v16h*)(bp + (size_t)(kt + 1) * 512);
                acc = __builtin_amdgcn_wmma_f32_16x16x32_f16(
                        false, aCur, false, bCur, (short)0, acc, false, false);
                aCur = aNxt; bCur = bNxt;
            }
            acc = __builtin_amdgcn_wmma_f32_16x16x32_f16(
                    false, aCur, false, bCur, (short)0, acc, false, false);

            // epilogue: D[m][n], m = r + (lane>=16 ? 8:0), n = nt*16 + (lane&15)
            int   n   = nt * 16 + (lane & 15);
            float b1n = b1[n];
            float w2n = W2[n];
            #pragma unroll
            for (int r = 0; r < 8; ++r) {
                float hv = acc[r] + b1n;
                float g  = 0.5f * hv * (1.0f + erff(hv * 0.70710678118654752f));
                scal[mlp][r] += g * w2n;
            }
        }
    }

    // ---- reduce over n within each 16-lane group (xor over bits 0..3)
    #pragma unroll
    for (int off = 1; off < 16; off <<= 1) {
        #pragma unroll
        for (int r = 0; r < 8; ++r) {
            scal[0][r] += __shfl_xor(scal[0][r], off, 32);
            scal[1][r] += __shfl_xor(scal[1][r], off, 32);
        }
    }
    if ((lane & 15) == 0) {
        int mh = (lane >> 4) << 3;                   // 0 or 8
        #pragma unroll
        for (int r = 0; r < 8; ++r) {
            sRed[wave][mh + r][0] = scal[0][r];
            sRed[wave][mh + r][1] = scal[1][r];
        }
    }
    __syncthreads();

    // ---- cross-wave reduce + scatter outputs
    if (tid < 16) {
        int m = tid;
        float E = 0.0f, F = 0.0f;
        #pragma unroll
        for (int w = 0; w < 8; ++w) { E += sRed[w][m][0]; F += sRed[w][m][1]; }
        E += b2e[0];
        F += b2f[0];
        int e = blk * 16 + m;
        int s = sSrc[m], d2 = sDst[m];
        float vx = pos[s * 3 + 0] - pos[d2 * 3 + 0];
        float vy = pos[s * 3 + 1] - pos[d2 * 3 + 1];
        float vz = pos[s * 3 + 2] - pos[d2 * 3 + 2];
        float nrm = sqrtf(vx * vx + vy * vy + vz * vz);
        float inv = 1.0f / fmaxf(nrm, 1e-12f);
        float* outF = out + B_;
        atomicAdd(&outF[s * 3 + 0], F * vx * inv);
        atomicAdd(&outF[s * 3 + 1], F * vy * inv);
        atomicAdd(&outF[s * 3 + 2], F * vz * inv);
        sE2[m] = E;
    }
    __syncthreads();
    if (tid == 0) {
        float sE = 0.0f;
        #pragma unroll
        for (int m = 0; m < 16; ++m) sE += sE2[m];
        int g = (blk * 16) / EPG_;
        atomicAdd(&out[g], sE);
    }
}

// ---------------------------------------------------------------- launch
extern "C" void kernel_launch(void* const* d_in, const int* in_sizes, int n_in,
                              void* d_out, int out_size, void* d_ws, size_t ws_size,
                              hipStream_t stream) {
    const float* x    = (const float*)d_in[0];
    const float* pos  = (const float*)d_in[1];
    // d_in[2] = batch (unused: graph id derivable), d_in[4,5] masks (static layout)
    const int*   ei   = (const int*)d_in[3];
    const float* W1e  = (const float*)d_in[6];
    const float* b1e  = (const float*)d_in[7];
    const float* W2e  = (const float*)d_in[8];
    const float* b2e  = (const float*)d_in[9];
    const float* W1f  = (const float*)d_in[10];
    const float* b1f  = (const float*)d_in[11];
    const float* W2f  = (const float*)d_in[12];
    const float* b2f  = (const float*)d_in[13];
    float*     out = (float*)d_out;
    _Float16*  Wpk = (_Float16*)d_ws;                // 2*1536*1024 halves = 6.29 MB

    zero_out_kernel<<<(out_size + 255) / 256, 256, 0, stream>>>(out, out_size);
    pack_w1_kernel<<<768, 256, 0, stream>>>(W1e, W1f, Wpk);
    edge_mlp_kernel<<<NE_ / 16, 256, 0, stream>>>(x, pos, ei, Wpk,
                                                  b1e, W2e, b2e,
                                                  b1f, W2f, b2f, out);
}
